// _StructuralEncoder_8899172237847
// MI455X (gfx1250) — compile-verified, matrix operand order checked by reading, not run on hardware
//
#include <hip/hip_runtime.h>

// ---------------------------------------------------------------------------
// CDNA5 / gfx1250 GAT (2-layer) implementation.
// Dense GEMMs use v_wmma_f32_16x16x32_bf16 (wave32, one 16x16 tile per wave).
// Sparse softmax-aggregation uses float-ordered atomicMax + atomicAdd; the
// 192MB L2 keeps the 51MB feature / accumulator buffers resident.
// ---------------------------------------------------------------------------

typedef __attribute__((ext_vector_type(16))) __bf16 v16bf;
typedef __attribute__((ext_vector_type(8)))  float  v8f;

union Frag16 { v16bf v; unsigned int u[8]; };

#define NEG_SLOPE 0.2f

__device__ __forceinline__ unsigned short f2bf(float f) {
  unsigned int u = __float_as_uint(f);
  u += 0x7fffu + ((u >> 16) & 1u);        // round-to-nearest-even
  return (unsigned short)(u >> 16);
}

// order-preserving float <-> uint encoding for atomicMax on floats
__device__ __forceinline__ unsigned int enc_f32(float f) {
  unsigned int u = __float_as_uint(f);
  return (u & 0x80000000u) ? ~u : (u | 0x80000000u);
}
__device__ __forceinline__ float dec_f32(unsigned int e) {
  unsigned int u = (e & 0x80000000u) ? (e & 0x7fffffffu) : ~e;
  return __uint_as_float(u);
}

__device__ __forceinline__ void edge_nodes(const int* __restrict__ ei, int E,
                                           int e, int& s, int& d) {
  if (e < E) { s = ei[e]; d = ei[E + e]; }  // edge_index[0,:], edge_index[1,:]
  else       { s = d = e - E; }             // appended self-loops
}

// ---------------------------------------------------------------------------
// elementwise conversions
// ---------------------------------------------------------------------------
__global__ void k_cvt_bf16(const float* __restrict__ in,
                           unsigned short* __restrict__ out, int n) {
  int i = blockIdx.x * blockDim.x + threadIdx.x;
  if (i < n) out[i] = f2bf(in[i]);
}

// W[K,M] f32 -> Wt[M,K] bf16 (K-contiguous rows for the B fragment)
__global__ void k_transpose_bf16(const float* __restrict__ W,
                                 unsigned short* __restrict__ Wt, int K, int M) {
  int i = blockIdx.x * blockDim.x + threadIdx.x;
  if (i >= K * M) return;
  int k = i / M, m = i % M;
  Wt[(size_t)m * K + k] = f2bf(W[i]);
}

// ---------------------------------------------------------------------------
// C[nrows,M] = A[nrows,K](bf16, row-major) x Bt[M,K](bf16, pre-transposed)
// One wave per 16x16 output tile. K-chunks of 32 feed
// v_wmma_f32_16x16x32_bf16. Fragment layout follows ISA 7.12.2:
//   A 16x32 bf16 : lane L -> M = L&15, half = L>>4;
//     VGPR j<4  holds K = half*8 + 2j(+1); VGPR j>=4 holds K = 16 + half*8 + ...
//   B mirrors A with lane = N index.  C/D: lane -> N=L&15, VGPR i -> M=(L>>4)*8+i
// ---------------------------------------------------------------------------
__global__ __launch_bounds__(256)
void k_wmma_gemm_bf16(const unsigned short* __restrict__ A,
                      const unsigned short* __restrict__ Bt,
                      float* __restrict__ C, int nrows, int K, int M) {
  const int lane   = threadIdx.x & 31;
  const int wave   = blockIdx.x * (blockDim.x >> 5) + (threadIdx.x >> 5);
  const int ctiles = M >> 4;
  const int tiles  = (nrows >> 4) * ctiles;
  if (wave >= tiles) return;
  const int rt = wave / ctiles, ct = wave % ctiles;
  const int ml = lane & 15, half = lane >> 4;

  const unsigned short* arow = A  + (size_t)(rt * 16 + ml) * K;
  const unsigned short* brow = Bt + (size_t)(ct * 16 + ml) * K;

  v8f c = {};
  for (int k0 = 0; k0 < K; k0 += 32) {
    Frag16 af, bf;
#pragma unroll
    for (int j = 0; j < 8; ++j) {
      int k = k0 + ((j < 4) ? (half * 8 + 2 * j) : (16 + half * 8 + 2 * (j - 4)));
      af.u[j] = *reinterpret_cast<const unsigned int*>(arow + k);
      bf.u[j] = *reinterpret_cast<const unsigned int*>(brow + k);
    }
    c = __builtin_amdgcn_wmma_f32_16x16x32_bf16(
        /*neg_a=*/false, af.v, /*neg_b=*/false, bf.v,
        /*c_mod=*/(short)0, c, /*reuse_a=*/false, /*reuse_b=*/false);
  }

  float* crow = C + (size_t)(rt * 16 + half * 8) * M + ct * 16 + ml;
#pragma unroll
  for (int i = 0; i < 8; ++i) crow[(size_t)i * M] = c[i];
}

// ---------------------------------------------------------------------------
// attention logits per node: als[n,h] = <H[n,h,:], a_src[h,:]>, same for ald
// ---------------------------------------------------------------------------
__global__ void k_att_coeff(const float* __restrict__ H,
                            const float* __restrict__ a_src,
                            const float* __restrict__ a_dst,
                            float* __restrict__ als, float* __restrict__ ald,
                            int n, int heads, int F) {
  int i = blockIdx.x * blockDim.x + threadIdx.x;
  if (i >= n * heads) return;
  int node = i / heads, h = i % heads;
  const float* hp = H + (size_t)node * heads * F + (size_t)h * F;
  const float* as = a_src + h * F;
  const float* ad = a_dst + h * F;
  float s = 0.f, d = 0.f;
  for (int f = 0; f < F; ++f) { float v = hp[f]; s += v * as[f]; d += v * ad[f]; }
  als[i] = s; ald[i] = d;
}

// segment max over destinations (ordered-uint atomicMax)
__global__ void k_edge_max(const int* __restrict__ ei, int E, int ET,
                           const float* __restrict__ als,
                           const float* __restrict__ ald,
                           unsigned int* __restrict__ mx, int heads) {
  int i = blockIdx.x * blockDim.x + threadIdx.x;
  if (i >= ET * heads) return;
  int e = i / heads, h = i % heads;
  int s, d; edge_nodes(ei, E, e, s, d);
  float l = als[s * heads + h] + ald[d * heads + h];
  l = (l > 0.f) ? l : NEG_SLOPE * l;
  atomicMax(&mx[d * heads + h], enc_f32(l));
}

// numerator exp(logit - max) per edge (stashed) + segment-sum denominator
__global__ void k_edge_exp(const int* __restrict__ ei, int E, int ET,
                           const float* __restrict__ als,
                           const float* __restrict__ ald,
                           const unsigned int* __restrict__ mx,
                           float* __restrict__ den, float* __restrict__ expo,
                           int heads) {
  int i = blockIdx.x * blockDim.x + threadIdx.x;
  if (i >= ET * heads) return;
  int e = i / heads, h = i % heads;
  int s, d; edge_nodes(ei, E, e, s, d);
  float l = als[s * heads + h] + ald[d * heads + h];
  l = (l > 0.f) ? l : NEG_SLOPE * l;
  float m = dec_f32(mx[d * heads + h]);
  if (!isfinite(m)) m = 0.f;
  float t = __expf(l - m);
  expo[(size_t)e * heads + h] = t;
  atomicAdd(&den[d * heads + h], t);
}

// out[dst, h*F+f] += H[src, h*F+f] * (expo / max(den,1e-16))
__global__ void k_edge_aggregate(const int* __restrict__ ei, int E, int ET,
                                 const float* __restrict__ H,
                                 const float* __restrict__ expo,
                                 const float* __restrict__ den,
                                 float* __restrict__ out, int heads, int F) {
  long long i = (long long)blockIdx.x * blockDim.x + threadIdx.x;
  const int HF = heads * F;
  if (i >= (long long)ET * HF) return;
  int e = (int)(i / HF), fid = (int)(i % HF);
  int h = fid / F;
  int s, d; edge_nodes(ei, E, e, s, d);
  float a = expo[(size_t)e * heads + h] / fmaxf(den[d * heads + h], 1e-16f);
  atomicAdd(&out[(size_t)d * HF + fid], H[(size_t)s * HF + fid] * a);
}

// v = elu(in + b[c]) -> bf16 (feeds GEMM2)
__global__ void k_bias_elu_bf16(const float* __restrict__ in,
                                const float* __restrict__ b,
                                unsigned short* __restrict__ outb, int n, int C) {
  int i = blockIdx.x * blockDim.x + threadIdx.x;
  if (i >= n) return;
  float v = in[i] + b[i % C];
  v = (v > 0.f) ? v : (__expf(v) - 1.f);
  outb[i] = f2bf(v);
}

// seed the output with the broadcast bias so aggregation adds on top
__global__ void k_init_bias(float* __restrict__ out, const float* __restrict__ b,
                            int n, int C) {
  int i = blockIdx.x * blockDim.x + threadIdx.x;
  if (i < n) out[i] = b[i % C];
}

// ---------------------------------------------------------------------------
extern "C" void kernel_launch(void* const* d_in, const int* in_sizes, int n_in,
                              void* d_out, int out_size, void* d_ws, size_t ws_size,
                              hipStream_t stream) {
  (void)n_in; (void)out_size; (void)ws_size;
  const float* x   = (const float*)d_in[0];
  const int*   ei  = (const int*)d_in[1];     // [2,E] int32 (jax x64 disabled)
  const float* W1  = (const float*)d_in[2];   // [128,256]
  const float* as1 = (const float*)d_in[3];   // [4,64]
  const float* ad1 = (const float*)d_in[4];
  const float* b1  = (const float*)d_in[5];   // [256]
  const float* W2  = (const float*)d_in[6];   // [256,64]
  const float* as2 = (const float*)d_in[7];   // [1,64]
  const float* ad2 = (const float*)d_in[8];
  const float* b2  = (const float*)d_in[9];   // [64]
  float* out = (float*)d_out;                 // [N,64]

  const int INC = 128, HC = 256, OC = 64, H1H = 4, FH = 64;
  const int N  = in_sizes[0] / INC;           // 50000
  const int E  = in_sizes[1] / 2;             // 800000
  const int ET = E + N;                       // + self loops

  // workspace layout (256B aligned)
  char* wsb = (char*)d_ws;
  size_t off = 0;
  auto take = [&](size_t bytes) {
    void* p = wsb + off;
    off = (off + bytes + 255) & ~(size_t)255;
    return p;
  };
  unsigned short* xb  = (unsigned short*)take((size_t)N * INC * 2);
  unsigned short* w1t = (unsigned short*)take((size_t)HC * INC * 2); // [256,128]
  unsigned short* w2t = (unsigned short*)take((size_t)OC * HC * 2);  // [64,256]
  float*          Hb  = (float*)take((size_t)N * HC * 4);            // H1, reused as H2
  float*          ag1 = (float*)take((size_t)N * HC * 4);            // layer-1 accumulator
  unsigned short* hmb = (unsigned short*)take((size_t)N * HC * 2);   // elu(out1+b1) bf16
  float*          als = (float*)take((size_t)N * H1H * 4);
  float*          ald = (float*)take((size_t)N * H1H * 4);
  unsigned int*   mx  = (unsigned int*)take((size_t)N * H1H * 4);
  float*          den = (float*)take((size_t)N * H1H * 4);
  float*          expo= (float*)take((size_t)ET * H1H * 4);

  const int TB = 256;
  // --- precision conversion / weight transpose ---
  k_cvt_bf16<<<(N * INC + TB - 1) / TB, TB, 0, stream>>>(x, xb, N * INC);
  k_transpose_bf16<<<(INC * HC + TB - 1) / TB, TB, 0, stream>>>(W1, w1t, INC, HC);
  k_transpose_bf16<<<(HC * OC + TB - 1) / TB, TB, 0, stream>>>(W2, w2t, HC, OC);

  // --- layer 1: H1 = x @ W1 ---
  {
    int tiles = (N / 16) * (HC / 16);
    k_wmma_gemm_bf16<<<(tiles + 7) / 8, 256, 0, stream>>>(xb, w1t, Hb, N, INC, HC);
  }
  k_att_coeff<<<(N * H1H + TB - 1) / TB, TB, 0, stream>>>(Hb, as1, ad1, als, ald, N, H1H, FH);
  hipMemsetAsync(mx,  0, (size_t)N * H1H * 4, stream);
  hipMemsetAsync(den, 0, (size_t)N * H1H * 4, stream);
  hipMemsetAsync(ag1, 0, (size_t)N * HC * 4, stream);
  k_edge_max<<<(ET * H1H + TB - 1) / TB, TB, 0, stream>>>(ei, E, ET, als, ald, mx, H1H);
  k_edge_exp<<<(ET * H1H + TB - 1) / TB, TB, 0, stream>>>(ei, E, ET, als, ald, mx, den, expo, H1H);
  {
    long long tot = (long long)ET * HC;
    k_edge_aggregate<<<(unsigned)((tot + TB - 1) / TB), TB, 0, stream>>>(
        ei, E, ET, Hb, expo, den, ag1, H1H, FH);
  }
  k_bias_elu_bf16<<<(N * HC + TB - 1) / TB, TB, 0, stream>>>(ag1, b1, hmb, N * HC, HC);

  // --- layer 2: H2 = elu(out1+b1) @ W2 (reuse Hb) ---
  {
    int tiles = (N / 16) * (OC / 16);
    k_wmma_gemm_bf16<<<(tiles + 7) / 8, 256, 0, stream>>>(hmb, w2t, Hb, N, HC, OC);
  }
  k_att_coeff<<<(N + TB - 1) / TB, TB, 0, stream>>>(Hb, as2, ad2, als, ald, N, 1, OC);
  hipMemsetAsync(mx,  0, (size_t)N * 4, stream);
  hipMemsetAsync(den, 0, (size_t)N * 4, stream);
  k_edge_max<<<(ET + TB - 1) / TB, TB, 0, stream>>>(ei, E, ET, als, ald, mx, 1);
  k_edge_exp<<<(ET + TB - 1) / TB, TB, 0, stream>>>(ei, E, ET, als, ald, mx, den, expo, 1);
  k_init_bias<<<(N * OC + TB - 1) / TB, TB, 0, stream>>>(out, b2, N * OC, OC);
  {
    long long tot = (long long)ET * OC;
    k_edge_aggregate<<<(unsigned)((tot + TB - 1) / TB), TB, 0, stream>>>(
        ei, E, ET, Hb, expo, den, out, 1, OC);
  }
}
